// CrossAttention_61332132987186
// MI455X (gfx1250) — compile-verified, matrix operand order checked by reading, not run on hardware
//
#include <hip/hip_runtime.h>
#include <hip/hip_bf16.h>

typedef __attribute__((ext_vector_type(16))) _Float16 v16h;
typedef __attribute__((ext_vector_type(8)))  float    v8f;
typedef __attribute__((ext_vector_type(4)))  uint32_t u32x4;
typedef __attribute__((ext_vector_type(8)))  uint32_t u32x8;

#define BATCH 8
#define CDIM  320
#define NTOK  4096
#define NHEAD 8
#define DH    40
#define CTX   77
#define CTXD  768
#define JP    80   // padded ctx length for k buffer (sim N-dim)
#define JPV   96   // padded ctx length for v^T buffer (AV K-dim, 3x32)
#define DPQ   64   // padded head dim for q/k (sim K-dim, 2x32)
#define DPV   48   // padded head dim for AV output (3x16 N-tiles)

// ---- workspace layout (bytes, all 256B aligned) ----
static const size_t OFF_WQT = 0;          // 320*320 f16   = 204800
static const size_t OFF_WOT = 204800;     // 320*320 f16   = 204800
static const size_t OFF_WKT = 409600;     // 320*768 f16   = 491520
static const size_t OFF_WVT = 901120;     // 320*768 f16   = 491520
static const size_t OFF_CTX = 1392640;    // 8*80*768 f16  = 983040
static const size_t OFF_Q   = 2375680;    // 8*8*4096*64 f16 = 33554432
static const size_t OFF_K   = 35930112;   // 8*8*80*64 f16 = 655360
static const size_t OFF_VT  = 36585472;   // 8*8*48*96 f16 = 589824
static const size_t OFF_AO  = 37175296;   // 8*4096*320 f16 = 20971520
static const int    ZERO_WORDS = (655360 + 589824) / 4; // k + vT regions (adjacent)

__device__ __forceinline__ v8f wmma16(v8f c, v16h a, v16h b) {
  // D = A(16x32 f16) x B(32x16 f16) + C(16x16 f32)
  return __builtin_amdgcn_wmma_f32_16x16x32_f16(false, a, false, b, (short)0, c,
                                                false, false);
}

// ---- Tensor Data Mover: 1-D contiguous f16 tile, global -> LDS ----
// D# per CDNA5 ISA ch.8: group0 = {count=1, lds_addr, global_addr, type=2},
// group1 = {data_size=1 (2B), tensor_dim0 = tile_dim0 = nelem, stride0 = nelem}.
// Tracked on TENSORcnt; 2-operand form (groups 2/3 NULL -> up-to-2D tensors).
__device__ __forceinline__ void tdm_load_1d(uint32_t lds_off, const void* gp,
                                            uint32_t nelem /* <= 65535 */) {
  uint64_t ga = (uint64_t)(uintptr_t)gp;
  u32x4 g0;
  g0[0] = 1u;                                       // count=1 (valid user D#)
  g0[1] = lds_off;                                  // lds_addr (bytes)
  g0[2] = (uint32_t)ga;                             // global_addr[31:0]
  g0[3] = (uint32_t)(ga >> 32) | (2u << 30);        // global_addr[56:32] | type=2
  u32x8 g1;
  g1[0] = (1u << 16);                               // data_size=1 -> 2 bytes/elem
  g1[1] = (nelem & 0xFFFFu) << 16;                  // tensor_dim0[15:0]
  g1[2] = (nelem >> 16) | (1u << 16);               // tensor_dim0[31:16], tensor_dim1=1
  g1[3] = (nelem & 0xFFFFu) << 16;                  // tile_dim0 = nelem
  g1[4] = 0u;                                       // tile_dim1/2 unused
  g1[5] = nelem;                                    // tensor_dim0_stride[31:0]
  g1[6] = 0u;                                       // stride0 hi / stride1 lo
  g1[7] = 0u;                                       // stride1 hi
  asm volatile("tensor_load_to_lds %0, %1" :: "s"(g0), "s"(g1) : "memory");
}

// -------------------- prep kernels --------------------

__global__ void k_zero(uint32_t* __restrict__ p, int n) {
  int i = blockIdx.x * blockDim.x + threadIdx.x;
  if (i < n) p[i] = 0u;
}

// transpose + f32->f16 for all four weight matrices ([N][K] layout so B-tile
// fragments are contiguous 32B loads per lane)
__global__ void k_convert_w(const float* __restrict__ Wq, const float* __restrict__ Wk,
                            const float* __restrict__ Wv, const float* __restrict__ Wo,
                            _Float16* __restrict__ WqT, _Float16* __restrict__ WkT,
                            _Float16* __restrict__ WvT, _Float16* __restrict__ WoT) {
  int i = blockIdx.x * 256 + threadIdx.x;
  if (i < 102400) {
    int n = i / 320, k = i % 320;
    WqT[n * 320 + k] = (_Float16)Wq[k * 320 + n];
  } else if (i < 204800) {
    int j = i - 102400; int n = j / 320, k = j % 320;
    WoT[n * 320 + k] = (_Float16)Wo[k * 320 + n];
  } else if (i < 450560) {
    int j = i - 204800; int n = j / 768, k = j % 768;
    WkT[n * 768 + k] = (_Float16)Wk[k * 320 + n];
  } else if (i < 696320) {
    int j = i - 450560; int n = j / 768, k = j % 768;
    WvT[n * 768 + k] = (_Float16)Wv[k * 320 + n];
  }
}

__global__ void k_convert_ctx(const float* __restrict__ ctx, _Float16* __restrict__ c16) {
  int i = blockIdx.x * 256 + threadIdx.x;
  if (i >= BATCH * JP * CTXD) return;
  int k = i % CTXD;
  int bj = i / CTXD;
  int j = bj % JP;
  int b = bj / JP;
  c16[i] = (j < CTX) ? (_Float16)ctx[((size_t)(b * CTX + j)) * CTXD + k] : (_Float16)0.f;
}

// -------------------- LayerNorm + Q projection --------------------
// one block = 16 tokens; 4 waves each compute 5 of the 20 N-tiles of q.
__global__ __launch_bounds__(128) void k_ln_q(const float* __restrict__ x,
                                              const float* __restrict__ gamma,
                                              const float* __restrict__ beta,
                                              const _Float16* __restrict__ WqT,
                                              _Float16* __restrict__ qb) {
  __shared__ __align__(32) _Float16 xn[16][CDIM];
  __shared__ float red[16][9], red2[16][9];
  __shared__ float mu_s[16], rs_s[16];
  int b  = blockIdx.x >> 8;
  int t0 = (blockIdx.x & 255) * 16;
  int tid = threadIdx.x;

  // zero the d = 40..63 pad of q for these 16 tokens (keeps sim K-loop exact)
  for (int i = tid; i < 16 * NHEAD * 24; i += 128) {
    int d  = 40 + (i % 24);
    int tk = (i / 24) & 15;
    int hh = i / (24 * 16);
    qb[(((size_t)(b * NHEAD + hh) * NTOK + t0 + tk) << 6) + d] = (_Float16)0.f;
  }

  // pass 1: mean / var over 320 channels, 8 partial slices per token
  int tk = tid & 15, sl = tid >> 4;
  float s = 0.f, s2 = 0.f;
  for (int i = 0; i < 40; ++i) {
    int c = sl * 40 + i;
    float v = x[((size_t)(b * CDIM + c) << 12) + t0 + tk];
    s += v; s2 += v * v;
  }
  red[tk][sl] = s; red2[tk][sl] = s2;
  __syncthreads();
  if (tid < 16) {
    float a = 0.f, a2 = 0.f;
    for (int k = 0; k < 8; ++k) { a += red[tid][k]; a2 += red2[tid][k]; }
    float mu  = a * (1.f / 320.f);
    float var = a2 * (1.f / 320.f) - mu * mu;
    mu_s[tid] = mu;
    rs_s[tid] = rsqrtf(var + 1e-5f);
  }
  __syncthreads();
  // pass 2: normalized f16 tile into LDS
  for (int i = tid; i < 16 * CDIM; i += 128) {
    int c = i >> 4, t = i & 15;
    float v = x[((size_t)(b * CDIM + c) << 12) + t0 + t];
    xn[t][c] = (_Float16)(((v - mu_s[t]) * rs_s[t]) * gamma[c] + beta[c]);
  }
  __syncthreads();

  int lane = tid & 31, wave = tid >> 5;
  int row = lane & 15, half = lane >> 4, mb = half * 8;
  for (int nt = wave; nt < 20; nt += 4) {
    v8f acc = {};
#pragma unroll
    for (int ks = 0; ks < 10; ++ks) {
      v16h a  = *(const v16h*)(&xn[row][ks * 32 + half * 16]);
      v16h bb = *(const v16h*)(WqT + (size_t)(nt * 16 + row) * CDIM + ks * 32 + half * 16);
      acc = wmma16(acc, a, bb);
    }
    int n = nt * 16 + row;
    int hh = n / DH, d = n % DH;
    size_t base = (size_t)(b * NHEAD + hh) * NTOK + t0 + mb;
#pragma unroll
    for (int r = 0; r < 8; ++r)
      qb[((base + r) << 6) + d] = (_Float16)acc[r];
  }
}

// -------------------- K / V projections --------------------
// one wave per (b, jtile, ntile, {k|v}); K-loop = 768/32 = 24 WMMAs.
__global__ __launch_bounds__(32) void k_kv(const _Float16* __restrict__ c16,
                                           const _Float16* __restrict__ WkT,
                                           const _Float16* __restrict__ WvT,
                                           _Float16* __restrict__ kb,
                                           _Float16* __restrict__ vb) {
  int bid = blockIdx.x;
  int isv = bid & 1;
  int nt  = (bid >> 1) % 20;
  int jt  = ((bid >> 1) / 20) % 5;
  int b   = (bid >> 1) / 100;
  int lane = threadIdx.x & 31;
  int row = lane & 15, half = lane >> 4, mb = half * 8;
  const _Float16* Wt = isv ? WvT : WkT;
  const _Float16* ap = c16 + (size_t)(b * JP + jt * 16) * CTXD;
  v8f acc = {};
#pragma unroll
  for (int ks = 0; ks < 24; ++ks) {
    v16h a  = *(const v16h*)(ap + (size_t)row * CTXD + ks * 32 + half * 16);
    v16h bb = *(const v16h*)(Wt + (size_t)(nt * 16 + row) * CTXD + ks * 32 + half * 16);
    acc = wmma16(acc, a, bb);
  }
  int n = nt * 16 + row;
  int hh = n / DH, d = n % DH;
  if (!isv) {
#pragma unroll
    for (int r = 0; r < 8; ++r) {
      int j = jt * 16 + mb + r;
      kb[((size_t)(b * NHEAD + hh) * JP + j) * DPQ + d] = (_Float16)acc[r];
    }
  } else {
#pragma unroll
    for (int r = 0; r < 8; ++r) {
      int j = jt * 16 + mb + r;
      vb[((size_t)(b * NHEAD + hh) * DPV + d) * JPV + j] = (_Float16)acc[r];
    }
  }
}

// -------------------- attention: QK^T + softmax + AV --------------------
// one block = 8 waves per (b, head, query-chunk); K and V tiles for the head
// are DMA'd into LDS once by the TDM, then every wave runs 4 query tiles
// entirely out of LDS.
__global__ __launch_bounds__(256) void k_attn(const _Float16* __restrict__ qb,
                                              const _Float16* __restrict__ kb,
                                              const _Float16* __restrict__ vb,
                                              _Float16* __restrict__ ao) {
  __shared__ __align__(32) _Float16 lk[JP][DPQ];      // 10240 B, K tile [j][d]
  __shared__ __align__(32) _Float16 lv[DPV][JPV];     // 9216 B, V^T tile [d][j]
  __shared__ __align__(32) _Float16 at[8][16][JPV];   // per-wave attn transpose
  int bid = blockIdx.x;
  int qc = bid & 7;          // query chunk (32 tiles each)
  int hh = (bid >> 3) & 7;
  int b  = bid >> 6;
  int tid = threadIdx.x;
  int lane = tid & 31, wave = tid >> 5;
  int row = lane & 15, half = lane >> 4, mb = half * 8;

  if (wave == 0) {
    // TDM: stage this head's K and V^T tiles (both fully contiguous)
    tdm_load_1d((uint32_t)(uintptr_t)&lk[0][0],
                kb + (size_t)(b * NHEAD + hh) * JP * DPQ, JP * DPQ);
    tdm_load_1d((uint32_t)(uintptr_t)&lv[0][0],
                vb + (size_t)(b * NHEAD + hh) * DPV * JPV, DPV * JPV);
    __builtin_amdgcn_s_wait_tensorcnt(0);
  }
  __syncthreads();

  // zero the j = 80..95 pad of this wave's attn scratch once
  for (int i = lane; i < 16 * 16; i += 32)
    at[wave][i >> 4][80 + (i & 15)] = (_Float16)0.f;

  const float scale = 0.15811388300841897f; // 1/sqrt(40)

  for (int it = 0; it < 4; ++it) {
    int qt = qc * 32 + wave * 4 + it;
    const _Float16* qp = qb + ((size_t)((b * NHEAD + hh) * NTOK + qt * 16)) * DPQ;

    v16h a0 = *(const v16h*)(qp + row * DPQ + half * 16);
    v16h a1 = *(const v16h*)(qp + row * DPQ + 32 + half * 16);

    v8f sim[5];
#pragma unroll
    for (int jt = 0; jt < 5; ++jt) {
      v8f acc = {};
      v16h b0 = *(const v16h*)(&lk[jt * 16 + row][half * 16]);
      acc = wmma16(acc, a0, b0);
      v16h b1 = *(const v16h*)(&lk[jt * 16 + row][32 + half * 16]);
      acc = wmma16(acc, a1, b1);
      sim[jt] = acc;
    }

#pragma unroll
    for (int jt = 0; jt < 5; ++jt) {
      int j = jt * 16 + row; // N index (context position) lives in lane%16
#pragma unroll
      for (int r = 0; r < 8; ++r)
        sim[jt][r] = (j < CTX) ? sim[jt][r] * scale : -1e30f;
    }

    // softmax: a C-layout row is one VGPR across a 16-lane half-group
#pragma unroll
    for (int r = 0; r < 8; ++r) {
      float m = sim[0][r];
#pragma unroll
      for (int jt = 1; jt < 5; ++jt) m = fmaxf(m, sim[jt][r]);
#pragma unroll
      for (int s = 1; s < 16; s <<= 1) m = fmaxf(m, __shfl_xor(m, s, 32));
      float sum = 0.f;
#pragma unroll
      for (int jt = 0; jt < 5; ++jt) {
        float e = __expf(sim[jt][r] - m);
        sim[jt][r] = e;
        sum += e;
      }
#pragma unroll
      for (int s = 1; s < 16; s <<= 1) sum += __shfl_xor(sum, s, 32);
      float inv = 1.f / sum;
#pragma unroll
      for (int jt = 0; jt < 5; ++jt) sim[jt][r] *= inv;
    }

    // transpose attn through LDS into A-fragment layout (per-wave scratch;
    // same-wave DS ops are in-order so no block barrier needed)
#pragma unroll
    for (int jt = 0; jt < 5; ++jt)
#pragma unroll
      for (int r = 0; r < 8; ++r)
        at[wave][mb + r][jt * 16 + row] = (_Float16)sim[jt][r];

    // AV: 3 N-tiles of head dim, K-loop = 96/32
#pragma unroll
    for (int dt = 0; dt < 3; ++dt) {
      v8f acc = {};
#pragma unroll
      for (int ks = 0; ks < 3; ++ks) {
        v16h a  = *(const v16h*)(&at[wave][row][ks * 32 + half * 16]);
        v16h bb = *(const v16h*)(&lv[dt * 16 + row][ks * 32 + half * 16]);
        acc = wmma16(acc, a, bb);
      }
      int d = dt * 16 + row;
      if (d < DH) {
#pragma unroll
        for (int r = 0; r < 8; ++r)
          ao[((size_t)(b * NTOK + qt * 16 + mb + r)) * CDIM + hh * DH + d] = (_Float16)acc[r];
      }
    }
  }
}

// -------------------- out projection + bias + residual --------------------
__global__ __launch_bounds__(128) void k_out(const _Float16* __restrict__ ao,
                                             const _Float16* __restrict__ WoT,
                                             const float* __restrict__ bout,
                                             const float* __restrict__ x,
                                             float* __restrict__ out) {
  __shared__ __align__(32) _Float16 at[16][CDIM]; // 16 contiguous rows = 10240 B
  int b  = blockIdx.x >> 8;
  int t0 = (blockIdx.x & 255) * 16;
  int tid = threadIdx.x;
  int lane = tid & 31, wave = tid >> 5;
  if (wave == 0) {
    // TDM: stage the 16-token A tile (fully contiguous in attn-out)
    tdm_load_1d((uint32_t)(uintptr_t)&at[0][0],
                ao + (size_t)(b * NTOK + t0) * CDIM, 16 * CDIM);
    __builtin_amdgcn_s_wait_tensorcnt(0);
  }
  __syncthreads();
  int row = lane & 15, half = lane >> 4, mb = half * 8;
  for (int nt = wave; nt < 20; nt += 4) {
    v8f acc = {};
#pragma unroll
    for (int ks = 0; ks < 10; ++ks) {
      v16h a  = *(const v16h*)(&at[row][ks * 32 + half * 16]);
      v16h bb = *(const v16h*)(WoT + (size_t)(nt * 16 + row) * CDIM + ks * 32 + half * 16);
      acc = wmma16(acc, a, bb);
    }
    int c = nt * 16 + row;
    float bia = bout[c];
    size_t base = ((size_t)(b * CDIM + c) << 12) + t0 + mb;
#pragma unroll
    for (int r = 0; r < 8; ++r)
      out[base + r] = acc[r] + bia + x[base + r]; // residual, [b,c,h,w] layout
  }
}

// -------------------- launch --------------------
extern "C" void kernel_launch(void* const* d_in, const int* in_sizes, int n_in,
                              void* d_out, int out_size, void* d_ws, size_t ws_size,
                              hipStream_t stream) {
  (void)in_sizes; (void)n_in; (void)out_size; (void)ws_size;
  const float* x     = (const float*)d_in[0];
  const float* ctx   = (const float*)d_in[1];
  const float* Wq    = (const float*)d_in[2];
  const float* Wk    = (const float*)d_in[3];
  const float* Wv    = (const float*)d_in[4];
  const float* Wo    = (const float*)d_in[5];
  const float* bout  = (const float*)d_in[6];
  const float* gamma = (const float*)d_in[7];
  const float* beta  = (const float*)d_in[8];
  float* out = (float*)d_out;

  char* ws = (char*)d_ws;
  _Float16* WqT = (_Float16*)(ws + OFF_WQT);
  _Float16* WoT = (_Float16*)(ws + OFF_WOT);
  _Float16* WkT = (_Float16*)(ws + OFF_WKT);
  _Float16* WvT = (_Float16*)(ws + OFF_WVT);
  _Float16* c16 = (_Float16*)(ws + OFF_CTX);
  _Float16* qb  = (_Float16*)(ws + OFF_Q);
  _Float16* kb  = (_Float16*)(ws + OFF_K);
  _Float16* vb  = (_Float16*)(ws + OFF_VT);
  _Float16* ao  = (_Float16*)(ws + OFF_AO);

  k_zero<<<(ZERO_WORDS + 255) / 256, 256, 0, stream>>>((uint32_t*)(ws + OFF_K), ZERO_WORDS);
  k_convert_w<<<(696320 + 255) / 256, 256, 0, stream>>>(Wq, Wk, Wv, Wo, WqT, WkT, WvT, WoT);
  k_convert_ctx<<<(BATCH * JP * CTXD + 255) / 256, 256, 0, stream>>>(ctx, c16);

  k_ln_q<<<BATCH * (NTOK / 16), 128, 0, stream>>>(x, gamma, beta, WqT, qb);
  k_kv<<<BATCH * 5 * 20 * 2, 32, 0, stream>>>(c16, WkT, WvT, kb, vb);
  k_attn<<<BATCH * NHEAD * 8, 256, 0, stream>>>(qb, kb, vb, ao);
  k_out<<<BATCH * (NTOK / 16), 128, 0, stream>>>(ao, WoT, bout, x, out);
}
